// MoonElecOut_34815004901944
// MI455X (gfx1250) — compile-verified, matrix operand order checked by reading, not run on hardware
//
#include <hip/hip_runtime.h>

#define N_EL 8192
#define N_NB 24
#define DIM  256
#define F0   32
#define F1   16
#define N_ENV 8
#define CUTOFF 5.0f

typedef __attribute__((ext_vector_type(16))) __bf16 v16bf;
typedef __attribute__((ext_vector_type(8)))  float  v8f;
typedef __attribute__((ext_vector_type(4)))  int    v4i;

struct Frag32 { v4i lo, hi; };

__device__ __forceinline__ v16bf make_frag(v4i lo, v4i hi) {
  Frag32 t; t.lo = lo; t.hi = hi;
  return __builtin_bit_cast(v16bf, t);
}

// fp32 -> bf16 with round-to-nearest-even (stored as raw ushort)
__device__ __forceinline__ unsigned short f2bf(float f) {
  unsigned int u = __float_as_uint(f);
  u += 0x7FFFu + ((u >> 16) & 1u);
  return (unsigned short)(u >> 16);
}

__device__ __forceinline__ float silu_f(float x) {
  return x / (1.0f + __expf(-x));
}

// ---------------------------------------------------------------------------
// Weight prep: fp32 [K=256][N=256] -> bf16 pre-swizzled WMMA B fragments.
// Fragment for (kt,nt): 32 lanes x 16 bf16 contiguous (1024B), lane order per
// ISA 16-bit layout: lane = nn (+16 when kk&8), vgpr v = (kk&16)>>2 | (kk&7)>>1,
// slot = kk&1.  Storage offset (ushorts): (nt*8+kt)*512 + lane*16 + v*2 + slot.
// ---------------------------------------------------------------------------
__global__ void __launch_bounds__(256)
prep_w256(const float* __restrict__ W, unsigned short* __restrict__ dst) {
  int idx = blockIdx.x * 256 + threadIdx.x;       // 65536 elements
  int k = idx >> 8, n = idx & 255;
  int kt = k >> 5, kk = k & 31, nt = n >> 4, nn = n & 15;
  int lane = nn + ((kk & 8) ? 16 : 0);
  int v = ((kk & 16) >> 2) | ((kk & 7) >> 1);
  int slot = kk & 1;
  dst[(nt * 8 + kt) * 512 + lane * 16 + v * 2 + slot] = f2bf(W[idx]);
}

// Stack Wsame (rows 0..15) over Wdiff (rows 16..31): one 32x256 B matrix.
__global__ void __launch_bounds__(256)
prep_wcat(const float* __restrict__ Wsame, const float* __restrict__ Wdiff,
          unsigned short* __restrict__ dst) {
  int idx = blockIdx.x * 256 + threadIdx.x;       // 8192 elements
  int k = idx >> 8, n = idx & 255;                // k in 0..31
  float val = (k < 16) ? Wsame[k * 256 + n] : Wdiff[(k - 16) * 256 + n];
  int kk = k, nt = n >> 4, nn = n & 15;
  int lane = nn + ((kk & 8) ? 16 : 0);
  int v = ((kk & 16) >> 2) | ((kk & 7) >> 1);
  int slot = kk & 1;
  dst[nt * 512 + lane * 16 + v * 2 + slot] = f2bf(val);
}

// ---------------------------------------------------------------------------
// Phase A: elec1 = silu(elec @ W_in + b_in).  16 rows/block, 8 waves x 2 tiles.
// ---------------------------------------------------------------------------
__global__ void __launch_bounds__(256)
k_elec1(const float* __restrict__ elec, const unsigned short* __restrict__ WinF,
        const float* __restrict__ b_in, float* __restrict__ elec1) {
  __shared__ unsigned short As[16 * 256];         // 8 KB bf16 A tile
  int r0 = blockIdx.x * 16;
  for (int idx = threadIdx.x; idx < 4096; idx += 256)
    As[idx] = f2bf(elec[r0 * 256 + idx]);
  __syncthreads();

  int lane = threadIdx.x & 31;
  int wave = threadIdx.x >> 5;
  int hl = (lane >> 4) & 1;                       // which K-half this lane holds
  int arow = lane & 15;

#pragma unroll
  for (int t = 0; t < 2; ++t) {
    int nt = wave * 2 + t;
    v8f acc = {};
#pragma unroll
    for (int kt = 0; kt < 8; ++kt) {
      int ab = arow * 256 + kt * 32 + hl * 8;
      v16bf a = make_frag(*(const v4i*)&As[ab], *(const v4i*)&As[ab + 16]);
      const v4i* bp = (const v4i*)(WinF + (nt * 8 + kt) * 512 + lane * 16);
      v16bf b = make_frag(bp[0], bp[1]);
      acc = __builtin_amdgcn_wmma_f32_16x16x32_bf16(false, a, false, b,
                                                    (short)0, acc, false, false);
    }
    int col = nt * 16 + (lane & 15);
    float bias = b_in[col];
#pragma unroll
    for (int i = 0; i < 8; ++i) {
      int row = (lane < 16) ? i : (i + 8);
      elec1[(r0 + row) * 256 + col] = silu_f(acc[i] + bias);
    }
  }
}

// ---------------------------------------------------------------------------
// Phase B: one block per electron.
//  1) 24 lanes compute beta[16] (4->32 MLP + env + cutoff + 40->16).
//  2) Spin-select GEMM: A[32x32] = [beta|0 , 0|beta], B = [Wsame;Wdiff] -> gamma.
//  3) hmsg[d] = sum_k silu(elec1[d]+hinit_nb[k][d]) * gamma[k][d]  (streams 24KB).
// ---------------------------------------------------------------------------
__global__ void __launch_bounds__(256)
k_pair(const float* __restrict__ r, const float* __restrict__ r_nb,
       const int* __restrict__ s, const int* __restrict__ s_nb,
       const float* __restrict__ hinit_nb,
       const float* __restrict__ ee_scales, const float* __restrict__ ee_kernel,
       const float* __restrict__ ee_bias, const float* __restrict__ Wf,
       const float* __restrict__ bfb, const unsigned short* __restrict__ WcatF,
       const float* __restrict__ elec1, float* __restrict__ hmsg) {
  __shared__ unsigned short Abeta[32 * 32];       // 2 KB  (padded A matrix)
  __shared__ float gammaS[24 * 256];              // 24 KB
  int n = blockIdx.x;
  int tid = threadIdx.x;

  if (tid < 32) {
    if (tid < 24) {
      int k = tid;
      float dx = r_nb[(n * 24 + k) * 3 + 0] - r[n * 3 + 0];
      float dy = r_nb[(n * 24 + k) * 3 + 1] - r[n * 3 + 1];
      float dz = r_nb[(n * 24 + k) * 3 + 2] - r[n * 3 + 2];
      float dist = sqrtf(dx * dx + dy * dy + dz * dz + 1e-12f);
      float feats[4] = {dist, dx, dy, dz};
      float h[40];
#pragma unroll
      for (int j = 0; j < 32; ++j) {
        float t = ee_bias[j];
#pragma unroll
        for (int i = 0; i < 4; ++i) t += feats[i] * ee_kernel[i * 32 + j];
        h[j] = silu_f(t);
      }
#pragma unroll
      for (int e = 0; e < 8; ++e) {
        float q = dist / ee_scales[e];
        h[32 + e] = __expf(-q * q);
      }
      float x = dist * (1.0f / CUTOFF);
      float cut = (dist < CUTOFF) ? (1.0f - x) * (1.0f - x) * (1.0f + 2.0f * x)
                                  : 0.0f;
      bool same = (s[n] == s_nb[n * 24 + k]);
#pragma unroll
      for (int j = 0; j < 16; ++j) {
        float bj = bfb[j];
        for (int i = 0; i < 40; ++i) bj += h[i] * Wf[i * 16 + j];
        bj *= cut;
        unsigned short bb = f2bf(bj);
        Abeta[k * 32 + j]      = same ? bb : (unsigned short)0;
        Abeta[k * 32 + 16 + j] = same ? (unsigned short)0 : bb;
      }
    } else {
#pragma unroll
      for (int j = 0; j < 32; ++j) Abeta[tid * 32 + j] = 0;
    }
  }
  __syncthreads();

  // gamma[24x256] = Abeta[32x32] @ Wcat[32x256]  (32 tiles / 8 waves)
  int lane = tid & 31, wave = tid >> 5;
  int hl = (lane >> 4) & 1;
#pragma unroll
  for (int t = 0; t < 4; ++t) {
    int tile = wave * 4 + t;
    int mt = tile >> 4, nt = tile & 15;
    int arow = mt * 16 + (lane & 15);
    int ab = arow * 32 + hl * 8;
    v16bf a = make_frag(*(const v4i*)&Abeta[ab], *(const v4i*)&Abeta[ab + 16]);
    const v4i* bp = (const v4i*)(WcatF + nt * 512 + lane * 16);
    v16bf b = make_frag(bp[0], bp[1]);
    v8f acc = {};
    acc = __builtin_amdgcn_wmma_f32_16x16x32_bf16(false, a, false, b,
                                                  (short)0, acc, false, false);
#pragma unroll
    for (int i = 0; i < 8; ++i) {
      int row = mt * 16 + ((lane < 16) ? i : (i + 8));
      if (row < 24) gammaS[row * 256 + nt * 16 + (lane & 15)] = acc[i];
    }
  }
  __syncthreads();

  // streaming silu-weighted reduction over neighbors (coalesced hinit_nb)
  int d = tid;
  float e1 = elec1[n * 256 + d];
  float acc = 0.0f;
#pragma unroll 4
  for (int k = 0; k < 24; ++k) {
    float hv = hinit_nb[(size_t)(n * 24 + k) * 256 + d];
    acc += silu_f(e1 + hv) * gammaS[k * 256 + d];
  }
  hmsg[n * 256 + d] = acc;
}

// ---------------------------------------------------------------------------
// Phase C: fused W1/W2/W3 chain.  Activations bounce through the LDS bf16
// A-buffer between stages; residual adds (hmsg, msg, elec1) in epilogues.
// ---------------------------------------------------------------------------
__global__ void __launch_bounds__(256)
k_out(const float* __restrict__ elec1, const float* __restrict__ hmsg,
      const float* __restrict__ msg,
      const unsigned short* __restrict__ W1F, const float* __restrict__ b1,
      const unsigned short* __restrict__ W2F, const float* __restrict__ b2,
      const unsigned short* __restrict__ W3F, const float* __restrict__ b3,
      const float* __restrict__ scale, float* __restrict__ out) {
  __shared__ unsigned short As[16 * 256];
  int r0 = blockIdx.x * 16;
  for (int idx = threadIdx.x; idx < 4096; idx += 256)
    As[idx] = f2bf(elec1[r0 * 256 + idx]);
  __syncthreads();

  int lane = threadIdx.x & 31, wave = threadIdx.x >> 5;
  int hl = (lane >> 4) & 1;
  int arow = lane & 15;
  float sc = scale[0];

  const unsigned short* Ws[3] = {W1F, W2F, W3F};
  const float* bs[3] = {b1, b2, b3};

#pragma unroll
  for (int stage = 0; stage < 3; ++stage) {
    v8f acc[2] = {{}, {}};
#pragma unroll
    for (int kt = 0; kt < 8; ++kt) {
      int ab = arow * 256 + kt * 32 + hl * 8;
      v16bf a = make_frag(*(const v4i*)&As[ab], *(const v4i*)&As[ab + 16]);
#pragma unroll
      for (int t = 0; t < 2; ++t) {
        int nt = wave * 2 + t;
        const v4i* bp = (const v4i*)(Ws[stage] + (nt * 8 + kt) * 512 + lane * 16);
        v16bf b = make_frag(bp[0], bp[1]);
        acc[t] = __builtin_amdgcn_wmma_f32_16x16x32_bf16(false, a, false, b,
                                                         (short)0, acc[t],
                                                         false, false);
      }
    }
    __syncthreads();                              // all As reads done
#pragma unroll
    for (int t = 0; t < 2; ++t) {
      int nt = wave * 2 + t;
      int col = nt * 16 + (lane & 15);
      float bias = bs[stage][col];
#pragma unroll
      for (int i = 0; i < 8; ++i) {
        int row = (lane < 16) ? i : (i + 8);
        int g = (r0 + row) * 256 + col;
        float val = acc[t][i] + bias;
        if (stage == 0) val += hmsg[g];
        else if (stage == 1) val += msg[g];
        float a2 = silu_f(val);
        if (stage == 2) out[g] = (a2 + elec1[g]) * sc;
        else            As[row * 256 + col] = f2bf(a2);
      }
    }
    __syncthreads();
  }
}

// ---------------------------------------------------------------------------
extern "C" void kernel_launch(void* const* d_in, const int* in_sizes, int n_in,
                              void* d_out, int out_size, void* d_ws, size_t ws_size,
                              hipStream_t stream) {
  const float* elec      = (const float*)d_in[0];
  const float* msg       = (const float*)d_in[1];
  const float* r         = (const float*)d_in[2];
  const float* r_nb      = (const float*)d_in[3];
  const int*   s         = (const int*)d_in[4];
  const int*   s_nb      = (const int*)d_in[5];
  const float* hinit_nb  = (const float*)d_in[6];
  const float* W_in      = (const float*)d_in[7];
  const float* b_in      = (const float*)d_in[8];
  const float* ee_scales = (const float*)d_in[9];
  const float* ee_kernel = (const float*)d_in[10];
  const float* ee_bias   = (const float*)d_in[11];
  const float* Wf        = (const float*)d_in[12];
  const float* bfb       = (const float*)d_in[13];
  const float* Wsame     = (const float*)d_in[14];
  const float* Wdiff     = (const float*)d_in[15];
  const float* W1        = (const float*)d_in[16];
  const float* b1        = (const float*)d_in[17];
  const float* W2        = (const float*)d_in[18];
  const float* b2        = (const float*)d_in[19];
  const float* W3        = (const float*)d_in[20];
  const float* b3        = (const float*)d_in[21];
  const float* scale     = (const float*)d_in[22];
  float* out = (float*)d_out;

  char* ws = (char*)d_ws;
  float* elec1 = (float*)ws;                                   // 8 MB
  float* hmsg  = (float*)(ws + (size_t)N_EL * DIM * 4);        // 8 MB
  unsigned short* WinF  = (unsigned short*)(ws + (size_t)2 * N_EL * DIM * 4);
  unsigned short* W1F   = WinF + 65536;
  unsigned short* W2F   = W1F + 65536;
  unsigned short* W3F   = W2F + 65536;
  unsigned short* WcatF = W3F + 65536;                         // 8192 ushorts

  prep_w256<<<256, 256, 0, stream>>>(W_in, WinF);
  prep_w256<<<256, 256, 0, stream>>>(W1, W1F);
  prep_w256<<<256, 256, 0, stream>>>(W2, W2F);
  prep_w256<<<256, 256, 0, stream>>>(W3, W3F);
  prep_wcat<<<32, 256, 0, stream>>>(Wsame, Wdiff, WcatF);

  k_elec1<<<N_EL / 16, 256, 0, stream>>>(elec, WinF, b_in, elec1);
  k_pair<<<N_EL, 256, 0, stream>>>(r, r_nb, s, s_nb, hinit_nb, ee_scales,
                                   ee_kernel, ee_bias, Wf, bfb, WcatF,
                                   elec1, hmsg);
  k_out<<<N_EL / 16, 256, 0, stream>>>(elec1, hmsg, msg, W1F, b1, W2F, b2,
                                       W3F, b3, scale, out);
}
